// MultiHeadAttentionAwareTemporalContex_qc_k1d_19765439496928
// MI455X (gfx1250) — matrix operand order repaired and checked by reading, not compiled, not run on hardware
//
#include <hip/hip_runtime.h>
#include <cstdint>

// Problem constants (B,N,T,D,H,K) = (8,512,64,128,8,3), DK=16
#define BB   8
#define NN   512
#define TT   64
#define DD   128
#define HH   8

typedef __attribute__((ext_vector_type(16))) _Float16 v16h;
typedef __attribute__((ext_vector_type(8)))  float    v8f;

union Frag {
    v16h     h;
    uint4    q[2];
    uint32_t u[8];
};

// f16 tile row stride: 128 cols + 8 pad = 136 halfs = 272 bytes (16B aligned)
#define ROWB   272
// score row stride: 64 f32 + pad = 68 f32 = 272 bytes
#define SROWB  272
// vT row stride: 64 + 8 = 72 halfs = 144 bytes (16B aligned)
#define VTROWB 144
// P row stride: 64 + 8 = 72 halfs = 144 bytes
#define PROWB  144

// LDS byte layout (dynamic shared):
//   SBUF  : 8 waves x 64 rows x 272B = 139264   (stage0/1: raw q/k/v tiles alias here)
//   QBUF  : 64 x 272 = 17408                    (stage>=PV: X tile aliases here)
//   KBUF  : 64 x 272 = 17408
//   VTBUF : 128 x 144 = 18432
//   PBUF  : 8 waves x 64 x 144 = 73728
#define SBUF_OFF  0
#define SBUF_PW   17408
#define RAWQ_OFF  0
#define RAWK_OFF  17408
#define RAWV_OFF  34816
#define QBUF_OFF  139264
#define KBUF_OFF  156672
#define VT_OFF    174080
#define PBUF_OFF  192512
#define XBUF_OFF  QBUF_OFF
#define LDS_TOTAL 266240

// Workspace (dwords): prepacked B-fragments, 256 dwords per 32x16 f16 tile
#define WS_BQ 0          //  96 tiles (12 kt x 8 ot)
#define WS_BK 24576      //  96 tiles
#define WS_BV 49152      //  32 tiles (4 kt x 8 ot)
#define WS_BO 57344      //  32 tiles
#define WS_TOTAL_DW 65536

// ---------------------------------------------------------------------------
// Pre-pack Wq/Wk (D,D,1,3) and Wv/Wo (D,D) into WMMA B-fragment order (f16):
// tile (kt,ot); lane l holds column n=l&15, K = (l>>4)*16 .. +15 packed 2/dword.
// ---------------------------------------------------------------------------
__global__ void prepack_weights(const float* __restrict__ Wq,
                                const float* __restrict__ Wk,
                                const float* __restrict__ Wv,
                                const float* __restrict__ Wo,
                                uint32_t* __restrict__ ws) {
    int gid = blockIdx.x * blockDim.x + threadIdx.x;
    if (gid >= WS_TOTAL_DW) return;
    int region, idx; const float* W;
    if      (gid < WS_BK) { region = 0; idx = gid;          W = Wq; }
    else if (gid < WS_BV) { region = 1; idx = gid - WS_BK;  W = Wk; }
    else if (gid < WS_BO) { region = 2; idx = gid - WS_BV;  W = Wv; }
    else                  { region = 3; idx = gid - WS_BO;  W = Wo; }
    int tile = idx >> 8;
    int lane = (idx >> 3) & 31;
    int j    = idx & 7;
    int n    = lane & 15;
    int K0   = ((lane >> 4) << 4) + 2 * j;   // even K index within 32-chunk
    int kt = tile >> 3, ot = tile & 7;
    int col = ot * 16 + n;                   // output channel (o or e)
    float f0, f1;
    if (region <= 1) {
        // B[kk, o] = W[o, i, k], kk = k*128 + i, kk-tile base = kt*32
        int kk0 = kt * 32 + K0;
        int kk1 = kk0 + 1;
        f0 = W[(col * 128 + (kk0 & 127)) * 3 + (kk0 >> 7)];
        f1 = W[(col * 128 + (kk1 & 127)) * 3 + (kk1 >> 7)];
    } else {
        // B[d, e] = W[e, d]
        int d0 = kt * 32 + K0;
        f0 = W[col * 128 + d0];
        f1 = W[col * 128 + d0 + 1];
    }
    union { _Float16 h[2]; uint32_t u; } pk;
    pk.h[0] = (_Float16)f0; pk.h[1] = (_Float16)f1;
    ws[gid] = pk.u;
}

// ---------------------------------------------------------------------------
// Fused kernel: one workgroup per (b,n); 256 threads = 8 waves; head h = wave.
// ---------------------------------------------------------------------------
__global__ __launch_bounds__(256, 1)
void fused_temporal_attn(const float* __restrict__ query,
                         const float* __restrict__ key,
                         const float* __restrict__ value,
                         const float* __restrict__ bq, const float* __restrict__ bk,
                         const float* __restrict__ bv, const float* __restrict__ bo,
                         const uint32_t* __restrict__ ws,
                         float* __restrict__ out) {
    extern __shared__ __align__(16) char smem[];
    const int tid  = threadIdx.x;
    const int lane = tid & 31;
    const int wv   = tid >> 5;                      // wave id == head id
    const size_t tileBase = (size_t)blockIdx.x * (TT * DD);

    const int m   = lane & 15;                      // fragment row / col
    const int kb2 = (lane >> 4) << 4;               // A-frag K sub-offset (0|16)

    // ---------- Stage 0: raw (64x128) f32 tiles -> f16 LDS ----------
    {
        const float4* srcs[3] = { (const float4*)(query + tileBase),
                                  (const float4*)(key   + tileBase),
                                  (const float4*)(value + tileBase) };
        const int offs[3] = { RAWQ_OFF, RAWK_OFF, RAWV_OFF };
        for (int s3 = 0; s3 < 3; ++s3) {
            char* dst = smem + offs[s3];
            for (int it = 0; it < 8; ++it) {
                int idx4 = tid + it * 256;           // 0..2047
                float4 f = srcs[s3][idx4];
                int flat = idx4 * 4;
                int t = flat >> 7, d = flat & 127;
                union { _Float16 h[4]; uint2 u2; } pk;
                pk.h[0] = (_Float16)f.x; pk.h[1] = (_Float16)f.y;
                pk.h[2] = (_Float16)f.z; pk.h[3] = (_Float16)f.w;
                *(uint2*)(dst + t * ROWB + d * 2) = pk.u2;
            }
        }
    }
    __syncthreads();

    // ---------- Stage 1: q = conv(query) [causal], k = conv(key) [same],
    //                     v = value*Wv^T (stored transposed) ----------
    for (int i = 0; i < 12; ++i) {
        int job = wv + 8 * i;                        // 96 jobs, ot == wv always
        int gm, tl;
        if      (job < 32) { gm = 0; tl = job;      }
        else if (job < 64) { gm = 1; tl = job - 32; }
        else               { gm = 2; tl = job - 64; }
        int mt = tl >> 3, ot = tl & 7;
        int nkt = (gm == 2) ? 4 : 12;
        int shift = (gm == 0) ? -2 : ((gm == 1) ? -1 : 0);
        const char* rawb = smem + (gm == 0 ? RAWQ_OFF : (gm == 1 ? RAWK_OFF : RAWV_OFF));
        const uint32_t* wsb = ws + (gm == 0 ? WS_BQ : (gm == 1 ? WS_BK : WS_BV));
        v8f acc = {};
        for (int kt = 0; kt < nkt; ++kt) {
            Frag a;
            int ibase = (gm == 2) ? kt * 32 : (kt & 3) * 32;
            int ktap  = (gm == 2) ? 0 : (kt >> 2);
            int tsrc  = mt * 16 + m + ktap + shift;  // conv tap row
            if (tsrc >= 0 && tsrc < TT) {
                const char* rp = rawb + tsrc * ROWB + (ibase + kb2) * 2;
                a.q[0] = *(const uint4*)rp;
                a.q[1] = *(const uint4*)(rp + 16);
            } else {
                a.q[0] = make_uint4(0, 0, 0, 0);
                a.q[1] = make_uint4(0, 0, 0, 0);
            }
            Frag b;
            const uint32_t* bp = wsb + ((size_t)(kt * 8 + ot) << 8) + (lane << 3);
            b.q[0] = *(const uint4*)bp;
            b.q[1] = *(const uint4*)(bp + 4);
            acc = __builtin_amdgcn_wmma_f32_16x16x32_f16(false, a.h, false, b.h,
                                                         (short)0, acc, false, false);
        }
        int col  = ot * 16 + m;
        float bias = (gm == 0 ? bq : (gm == 1 ? bk : bv))[col];
        int trow = mt * 16 + ((lane >> 4) << 3);
        if (gm == 2) {
            // store v transposed: vT[e][t], 8 consecutive t -> one b128
            union { _Float16 h[8]; uint4 u4; } pk;
            for (int r = 0; r < 8; ++r) pk.h[r] = (_Float16)(acc[r] + bias);
            *(uint4*)(smem + VT_OFF + col * VTROWB + trow * 2) = pk.u4;
        } else {
            char* qk = smem + (gm == 0 ? QBUF_OFF : KBUF_OFF);
            for (int r = 0; r < 8; ++r)
                *(_Float16*)(qk + (trow + r) * ROWB + col * 2) = (_Float16)(acc[r] + bias);
        }
    }
    __syncthreads();

    // ---------- Stage 2a: scores S = q_h k_h^T / 4, causal mask ----------
    {
        char* sbuf = smem + SBUF_OFF + wv * SBUF_PW;
        const char* qb = smem + QBUF_OFF;
        const char* kbuf = smem + KBUF_OFF;
        for (int tt = 0; tt < 4; ++tt) {
            Frag a;                                   // DK=16, pad K 16..31 = 0
            const char* rp = qb + (tt * 16 + m) * ROWB + (wv * 16 + kb2) * 2;
            a.q[0] = (lane < 16) ? *(const uint4*)rp : make_uint4(0, 0, 0, 0);
            a.q[1] = (lane < 16) ? *(const uint4*)(rp + 16) : make_uint4(0, 0, 0, 0);
            for (int st = 0; st < 4; ++st) {
                Frag b;
                if (lane < 16) {                      // B rows K 0..15 real
                    const char* kp = kbuf + (st * 16 + m) * ROWB + (wv * 16) * 2;
                    b.q[0] = *(const uint4*)kp;
                    b.q[1] = *(const uint4*)(kp + 16);
                } else {                              // K 16..31 padded zero
                    b.q[0] = make_uint4(0, 0, 0, 0);
                    b.q[1] = make_uint4(0, 0, 0, 0);
                }
                v8f acc = {};
                acc = __builtin_amdgcn_wmma_f32_16x16x32_f16(false, a.h, false, b.h,
                                                             (short)0, acc, false, false);
                int scol = st * 16 + m;
                int trow = tt * 16 + ((lane >> 4) << 3);
                for (int r = 0; r < 8; ++r) {
                    int t = trow + r;
                    float v = acc[r] * 0.25f;         // 1/sqrt(DK)
                    if (scol > t) v = -1.0e9f;        // causal mask
                    *(float*)(sbuf + t * SROWB + scol * 4) = v;
                }
            }
        }
    }
    __syncthreads();

    // ---------- Stage 2b: row softmax, write P (f16) ----------
    {
        const char* sbuf = smem + SBUF_OFF + wv * SBUF_PW;
        char* pbuf = smem + PBUF_OFF + wv * (64 * PROWB);
        for (int half = 0; half < 2; ++half) {
            int t = lane + half * 32;
            const float* row = (const float*)(sbuf + t * SROWB);
            float mx = -1.0e30f;
            for (int s = 0; s < TT; ++s) mx = fmaxf(mx, row[s]);
            float sum = 0.0f;
            for (int s = 0; s < TT; ++s) sum += __expf(row[s] - mx);
            float inv = 1.0f / sum;
            _Float16* prow = (_Float16*)(pbuf + t * PROWB);
            for (int s = 0; s < TT; ++s)
                prow[s] = (_Float16)(__expf(row[s] - mx) * inv);
        }
    }
    __syncthreads();   // also fences XBUF aliasing of QBUF

    // ---------- Stage 2c: X_h = P V_h ----------
    {
        const char* pbuf = smem + PBUF_OFF + wv * (64 * PROWB);
        const char* vt = smem + VT_OFF;
        char* xb = smem + XBUF_OFF;
        int e = wv * 16 + m;
        for (int mt = 0; mt < 4; ++mt) {
            v8f acc = {};
            for (int kt = 0; kt < 2; ++kt) {          // K = 64
                Frag a;
                const char* rp = pbuf + (mt * 16 + m) * PROWB + (kt * 32 + kb2) * 2;
                a.q[0] = *(const uint4*)rp;
                a.q[1] = *(const uint4*)(rp + 16);
                Frag b;                               // from vT: row e, cols s
                const char* vp = vt + e * VTROWB + (kt * 32 + ((lane >> 4) << 4)) * 2;
                b.q[0] = *(const uint4*)vp;
                b.q[1] = *(const uint4*)(vp + 16);
                acc = __builtin_amdgcn_wmma_f32_16x16x32_f16(false, a.h, false, b.h,
                                                             (short)0, acc, false, false);
            }
            int trow = mt * 16 + ((lane >> 4) << 3);
            for (int r = 0; r < 8; ++r)
                *(_Float16*)(xb + (trow + r) * ROWB + e * 2) = (_Float16)acc[r];
        }
    }
    __syncthreads();

    // ---------- Stage 3: out = X Wo^T + bo (f32 to global) ----------
    {
        const char* xb = smem + XBUF_OFF;
        float* og = out + tileBase;
        for (int i = 0; i < 4; ++i) {
            int tl = wv + 8 * i;
            int mt = tl >> 3, ot = tl & 7;
            v8f acc = {};
            for (int kt = 0; kt < 4; ++kt) {          // K = 128
                Frag a;
                const char* rp = xb + (mt * 16 + m) * ROWB + (kt * 32 + kb2) * 2;
                a.q[0] = *(const uint4*)rp;
                a.q[1] = *(const uint4*)(rp + 16);
                Frag b;
                const uint32_t* bp = ws + WS_BO + ((size_t)(kt * 8 + ot) << 8) + (lane << 3);
                b.q[0] = *(const uint4*)bp;
                b.q[1] = *(const uint4*)(bp + 4);
                acc = __builtin_amdgcn_wmma_f32_16x16x32_f16(false, a.h, false, b.h,
                                                             (short)0, acc, false, false);
            }
            int e = ot * 16 + m;
            float bias = bo[e];
            int trow = mt * 16 + ((lane >> 4) << 3);
            for (int r = 0; r < 8; ++r)
                og[(size_t)(trow + r) * DD + e] = acc[r] + bias;
        }
    }
}

extern "C" void kernel_launch(void* const* d_in, const int* in_sizes, int n_in,
                              void* d_out, int out_size, void* d_ws, size_t ws_size,
                              hipStream_t stream) {
    const float* query = (const float*)d_in[0];
    const float* key   = (const float*)d_in[1];
    const float* value = (const float*)d_in[2];
    // d_in[3] = mask (tril, hardcoded causal in kernel)
    const float* Wq = (const float*)d_in[4];
    const float* bq = (const float*)d_in[5];
    const float* Wk = (const float*)d_in[6];
    const float* bk = (const float*)d_in[7];
    const float* Wv = (const float*)d_in[8];
    const float* bv = (const float*)d_in[9];
    const float* Wo = (const float*)d_in[10];
    const float* bo = (const float*)d_in[11];
    float*    out = (float*)d_out;
    uint32_t* ws  = (uint32_t*)d_ws;

    (void)in_sizes; (void)n_in; (void)out_size; (void)ws_size;

    prepack_weights<<<WS_TOTAL_DW / 256, 256, 0, stream>>>(Wq, Wk, Wv, Wo, ws);

    hipFuncSetAttribute((const void*)fused_temporal_attn,
                        hipFuncAttributeMaxDynamicSharedMemorySize, LDS_TOTAL);
    fused_temporal_attn<<<BB * NN, 256, LDS_TOTAL, stream>>>(
        query, key, value, bq, bk, bv, bo, ws, out);
}
